// TinyViTBlock_2370821947976
// MI455X (gfx1250) — compile-verified, matrix operand order checked by reading, not run on hardware
//
#include <hip/hip_runtime.h>
#include <hip/hip_bf16.h>

typedef _Float16 half8  __attribute__((ext_vector_type(8)));
typedef _Float16 half16 __attribute__((ext_vector_type(16)));
typedef float    float8 __attribute__((ext_vector_type(8)));

#define C_      384
#define NH_     12
#define KD_     32
#define NTOK    49
#define NWIN    1024
#define MROWS   50176   /* 1024*49 == 64*784 */
#define QKVO    1152
#define HID_    1536
#define LL      784
#define ATT_SCALE 0.17677669529663689f

#if defined(__gfx1250__) && __has_builtin(__builtin_amdgcn_global_load_async_to_lds_b128)
#define HAVE_ASYNC_LDS 1
#else
#define HAVE_ASYNC_LDS 0
#endif

__device__ __forceinline__ float8 f8zero() {
  float8 z = {0.f,0.f,0.f,0.f,0.f,0.f,0.f,0.f};
  return z;
}

// Load a 16x32 f16 operand fragment (A-style: row per lane; B-style: column per
// lane reading the [N][K] weight tile) per ISA 7.12.2 layout.
__device__ __forceinline__ half16 frag_ld(const _Float16* base, int row0, int stride,
                                          int k0, int lane) {
  int r  = row0 + (lane & 15);
  int kb = k0 + ((lane & 16) ? 8 : 0);
  const _Float16* p = base + r * stride + kb;
  half16 f;
#pragma unroll
  for (int i = 0; i < 8; ++i) f[i] = p[i];
#pragma unroll
  for (int i = 0; i < 8; ++i) f[8 + i] = p[16 + i];
  return f;
}

__device__ __forceinline__ float8 wmma_f16(half16 a, half16 b, float8 c) {
  return __builtin_amdgcn_wmma_f32_16x16x32_f16(false, a, false, b, (short)0, c,
                                                false, false);
}

// Stage 16 consecutive halves (32B) global -> LDS, async if available.
#if HAVE_ASYNC_LDS
typedef int v4i_gcc __attribute__((vector_size(16)));
typedef __attribute__((address_space(1))) v4i_gcc glb_v4i;
typedef __attribute__((address_space(3))) v4i_gcc lds_v4i;
__device__ __forceinline__ void stage16(const _Float16* g, _Float16* l) {
  __builtin_amdgcn_global_load_async_to_lds_b128((glb_v4i*)g, (lds_v4i*)l, 0, 0);
  __builtin_amdgcn_global_load_async_to_lds_b128((glb_v4i*)g, (lds_v4i*)l, 16, 0);
}
#else
__device__ __forceinline__ void stage16(const _Float16* g, _Float16* l) {
  *(half8*)l       = *(const half8*)g;
  *((half8*)l + 1) = *((const half8*)g + 1);
}
#endif

__device__ __forceinline__ void stage_wait() {
#if HAVE_ASYNC_LDS
#if __has_builtin(__builtin_amdgcn_s_wait_asynccnt)
  __builtin_amdgcn_s_wait_asynccnt(0);
#else
  asm volatile("s_wait_asynccnt 0" ::: "memory");
#endif
#endif
}

// ---------------------------------------------------------------- utilities
__global__ __launch_bounds__(256) void f32_to_f16_kernel(const float* __restrict__ in,
                                                         _Float16* __restrict__ out, int n) {
  int i = blockIdx.x * 256 + threadIdx.x;
  if (i < n) out[i] = (_Float16)in[i];
}

__global__ __launch_bounds__(256) void bias_gather_kernel(const float* __restrict__ ab,
                                                          const int* __restrict__ idxs,
                                                          float* __restrict__ out) {
  int h = blockIdx.x;
  int i = blockIdx.y * 256 + threadIdx.x;
  if (i < NTOK * NTOK) out[h * NTOK * NTOK + i] = ab[h * NTOK + idxs[i]];
}

// --------------------------------------------------- LayerNorm (f32 -> f16)
// One wave per row (C=384 = 12 elems/lane), 8 rows per 256-thread block.
template <bool WINDOWED>
__global__ __launch_bounds__(256) void ln_kernel(const float* __restrict__ X,
                                                 const float* __restrict__ g,
                                                 const float* __restrict__ bb,
                                                 _Float16* __restrict__ out) {
  int wid = threadIdx.x >> 5, lane = threadIdx.x & 31;
  int r = blockIdx.x * 8 + wid;
  const float* src;
  if (WINDOWED) {
    int w = r / NTOK, t = r - w * NTOK;
    int b = w >> 4, wr = w & 15, wy = wr >> 2, wx = wr & 3;
    int i = t / 7, j = t - i * 7;
    int l = (wy * 7 + i) * 28 + wx * 7 + j;
    src = X + (size_t)(b * LL + l) * C_;
  } else {
    src = X + (size_t)r * C_;
  }
  float vals[12];
  float s = 0.f;
#pragma unroll
  for (int i = 0; i < 12; ++i) { vals[i] = src[lane + i * 32]; s += vals[i]; }
#pragma unroll
  for (int o = 16; o > 0; o >>= 1) s += __shfl_xor(s, o, 32);
  float mean = s * (1.f / 384.f);
  float v = 0.f;
#pragma unroll
  for (int i = 0; i < 12; ++i) { float d = vals[i] - mean; v += d * d; }
#pragma unroll
  for (int o = 16; o > 0; o >>= 1) v += __shfl_xor(v, o, 32);
  float rs = rsqrtf(v * (1.f / 384.f) + 1e-5f);
#pragma unroll
  for (int i = 0; i < 12; ++i) {
    int c = lane + i * 32;
    out[(size_t)r * C_ + c] = (_Float16)((vals[i] - mean) * rs * g[c] + bb[c]);
  }
}

// --------------------------------------------------------- WMMA tiled GEMM
// A[M,K] f16 row-major, W[N,K] f16 row-major (i.e. columns of B=W^T).
// Block tile 128x128, wave tile 32x64 (2x4 WMMA 16x16x32), K staged via LDS.
// EPI: 0 = f16 store (+bias)        1 = f16 store with exact GELU (+bias)
//      2 = f32 residual store with window->image permute (proj epilogue)
//      3 = f32 residual store, direct layout (fc2 epilogue -> d_out)
template <int EPI>
__device__ __forceinline__ void store_tile(float8 c, int mrow, int ncol,
                                           const float* __restrict__ bias,
                                           const float* __restrict__ res,
                                           float* __restrict__ outf,
                                           _Float16* __restrict__ outh,
                                           int N, int lane) {
  int r0 = mrow + ((lane & 16) ? 8 : 0);
  int n  = ncol + (lane & 15);
  float bv = bias[n];
#pragma unroll
  for (int r = 0; r < 8; ++r) {
    int m = r0 + r;
    float v = c[r] + bv;
    if constexpr (EPI == 0) {
      outh[(size_t)m * N + n] = (_Float16)v;
    } else if constexpr (EPI == 1) {
      float gg = 0.5f * v * (1.f + erff(v * 0.70710678118654752f));
      outh[(size_t)m * N + n] = (_Float16)gg;
    } else if constexpr (EPI == 2) {
      int w = m / NTOK, t = m - w * NTOK;
      int b = w >> 4, wr = w & 15, wy = wr >> 2, wx = wr & 3;
      int i = t / 7, j = t - i * 7;
      int l = (wy * 7 + i) * 28 + wx * 7 + j;
      size_t di = (size_t)(b * LL + l) * C_ + n;
      outf[di] = res[di] + v;
    } else {
      size_t di = (size_t)m * C_ + n;
      outf[di] = res[di] + v;
    }
  }
}

template <int EPI>
__global__ __launch_bounds__(256) void gemm_wmma_kernel(const _Float16* __restrict__ A,
                                                        const _Float16* __restrict__ W,
                                                        const float* __restrict__ bias,
                                                        const float* __restrict__ res,
                                                        float* __restrict__ outf,
                                                        _Float16* __restrict__ outh,
                                                        int M, int N, int K) {
  __shared__ _Float16 As[128 * 40];
  __shared__ _Float16 Bs[128 * 40];
  int tid = threadIdx.x, lane = tid & 31, wid = tid >> 5;
  int wm = wid & 3, wn = wid >> 2;       // 4 x 2 wave grid -> 32x64 per wave
  int m0 = blockIdx.y * 128;
  int n0 = blockIdx.x * 128;

  float8 acc[2][4];
#pragma unroll
  for (int i = 0; i < 2; ++i)
#pragma unroll
    for (int j = 0; j < 4; ++j) acc[i][j] = f8zero();

  int srow = tid >> 1, scol = (tid & 1) * 16;   // 128x32 staging pattern

  for (int k0 = 0; k0 < K; k0 += 32) {
    __syncthreads();
    // stage A tile 128x32 and B tile 128x32 (rows of W = columns of B)
    stage16(A + (size_t)(m0 + srow) * K + k0 + scol, As + srow * 40 + scol);
    stage16(W + (size_t)(n0 + srow) * K + k0 + scol, Bs + srow * 40 + scol);
    if (k0 + 32 < K) {
      __builtin_prefetch(A + (size_t)(m0 + srow) * K + k0 + 32 + scol, 0, 1);
      __builtin_prefetch(W + (size_t)(n0 + srow) * K + k0 + 32 + scol, 0, 1);
    }
    stage_wait();
    __syncthreads();

    half16 a0 = frag_ld(As, wm * 32,      40, 0, lane);
    half16 a1 = frag_ld(As, wm * 32 + 16, 40, 0, lane);
    half16 b[4];
#pragma unroll
    for (int j = 0; j < 4; ++j) b[j] = frag_ld(Bs, wn * 64 + j * 16, 40, 0, lane);

#pragma unroll
    for (int j = 0; j < 4; ++j) {
      acc[0][j] = wmma_f16(a0, b[j], acc[0][j]);
      acc[1][j] = wmma_f16(a1, b[j], acc[1][j]);
    }
  }

  int mrow = m0 + wm * 32, ncol = n0 + wn * 64;
#pragma unroll
  for (int i = 0; i < 2; ++i)
#pragma unroll
    for (int j = 0; j < 4; ++j)
      store_tile<EPI>(acc[i][j], mrow + i * 16, ncol + j * 16, bias, res, outf,
                      outh, N, lane);
}

// ------------------------------------------------- windowed attention (WMMA)
// One block (4 waves / 128 thr) per (window, head). 49 padded to 64.
#define SQ 40
#define SP 72
#define SV 72
#define SS 65
__global__ __launch_bounds__(128) void attn_kernel(const _Float16* __restrict__ qkv,
                                                   const float* __restrict__ bias12,
                                                   _Float16* __restrict__ O) {
  int h = blockIdx.x, w = blockIdx.y;
  int tid = threadIdx.x, lane = tid & 31, wid = tid >> 5;
  __shared__ _Float16 qs[64 * SQ];
  __shared__ _Float16 ks[64 * SQ];
  __shared__ _Float16 vsT[32 * SV];
  __shared__ _Float16 Ps[64 * SP];
  __shared__ float    Ss[64 * SS];

  const _Float16* src = qkv + (size_t)(w * NTOK) * QKVO + h * 96;

  for (int i = tid; i < 32 * SV; i += 128) vsT[i] = (_Float16)0.f;
  for (int i = tid; i < 64 * 32; i += 128) {
    int t = i >> 5, d = i & 31;
    _Float16 qv = (_Float16)0.f, kv = (_Float16)0.f;
    if (t < NTOK) {
      qv = src[(size_t)t * QKVO + d];
      kv = src[(size_t)t * QKVO + 32 + d];
    }
    qs[t * SQ + d] = qv;
    ks[t * SQ + d] = kv;
  }
  for (int i = tid; i < NTOK * 32; i += 128) {
    int t = i >> 5, d = i & 31;
    vsT[d * SV + t] = src[(size_t)t * QKVO + 64 + d];
  }
  __syncthreads();

  // S = q @ k^T  (each wave: one 16-row strip, 4 column tiles, K=32)
  {
    half16 aq = frag_ld(qs, wid * 16, SQ, 0, lane);
#pragma unroll
    for (int j = 0; j < 4; ++j) {
      half16 bk = frag_ld(ks, j * 16, SQ, 0, lane);
      float8 c = wmma_f16(aq, bk, f8zero());
      int r0 = wid * 16 + ((lane & 16) ? 8 : 0);
      int c0 = j * 16 + (lane & 15);
#pragma unroll
      for (int r = 0; r < 8; ++r) Ss[(r0 + r) * SS + c0] = c[r];
    }
  }
  __syncthreads();

  // softmax(S*scale + bias) rows, one thread per row
  if (tid < 64) {
    int t = tid;
    if (t < NTOK) {
      const float* bb = bias12 + (size_t)h * (NTOK * NTOK) + t * NTOK;
      float mx = -1e30f;
      for (int m = 0; m < NTOK; ++m) {
        float s = Ss[t * SS + m] * ATT_SCALE + bb[m];
        Ss[t * SS + m] = s;
        mx = fmaxf(mx, s);
      }
      float sum = 0.f;
      for (int m = 0; m < NTOK; ++m) {
        float e = __expf(Ss[t * SS + m] - mx);
        Ss[t * SS + m] = e;
        sum += e;
      }
      float inv = 1.f / sum;
      for (int m = 0; m < NTOK; ++m) Ps[t * SP + m] = (_Float16)(Ss[t * SS + m] * inv);
      for (int m = NTOK; m < 64; ++m) Ps[t * SP + m] = (_Float16)0.f;
    } else {
      for (int m = 0; m < 64; ++m) Ps[t * SP + m] = (_Float16)0.f;
    }
  }
  __syncthreads();

  // O = P @ V (K=64 -> two accumulated WMMAs; N=32 -> two column tiles)
  {
    half16 a0 = frag_ld(Ps, wid * 16, SP, 0, lane);
    half16 a1 = frag_ld(Ps, wid * 16, SP, 32, lane);
#pragma unroll
    for (int jn = 0; jn < 2; ++jn) {
      half16 b0 = frag_ld(vsT, jn * 16, SV, 0, lane);
      half16 b1 = frag_ld(vsT, jn * 16, SV, 32, lane);
      float8 c = wmma_f16(a1, b1, wmma_f16(a0, b0, f8zero()));
      int r0 = wid * 16 + ((lane & 16) ? 8 : 0);
      int c0 = jn * 16 + (lane & 15);
#pragma unroll
      for (int r = 0; r < 8; ++r) {
        int t = r0 + r;
        if (t < NTOK)
          O[(size_t)(w * NTOK + t) * C_ + h * KD_ + c0] = (_Float16)c[r];
      }
    }
  }
}

// ------------------------------------------- depthwise 3x3 conv + BatchNorm
__global__ __launch_bounds__(384) void dwconv_bn_kernel(const float* __restrict__ x1,
                                                        const float* __restrict__ cw,
                                                        const float* __restrict__ bg,
                                                        const float* __restrict__ bb,
                                                        const float* __restrict__ bm,
                                                        const float* __restrict__ bv,
                                                        float* __restrict__ x2) {
  int bl = blockIdx.x;
  int b = bl / LL, l = bl - b * LL;
  int y = l / 28, x = l - y * 28;
  int c = threadIdx.x;
  float acc = 0.f;
#pragma unroll
  for (int ky = 0; ky < 3; ++ky) {
    int yy = y + ky - 1;
    if (yy < 0 || yy >= 28) continue;
#pragma unroll
    for (int kx = 0; kx < 3; ++kx) {
      int xx = x + kx - 1;
      if (xx < 0 || xx >= 28) continue;
      acc += cw[c * 9 + ky * 3 + kx] *
             x1[((size_t)b * LL + yy * 28 + xx) * C_ + c];
    }
  }
  float inv = rsqrtf(bv[c] + 1e-5f);
  x2[((size_t)b * LL + l) * C_ + c] = (acc - bm[c]) * bg[c] * inv + bb[c];
}

// ----------------------------------------------------------------- launcher
extern "C" void kernel_launch(void* const* d_in, const int* in_sizes, int n_in,
                              void* d_out, int out_size, void* d_ws, size_t ws_size,
                              hipStream_t stream) {
  const float* x        = (const float*)d_in[0];
  const float* ln_a_g   = (const float*)d_in[1];
  const float* ln_a_b   = (const float*)d_in[2];
  const float* qkv_w    = (const float*)d_in[3];
  const float* qkv_b    = (const float*)d_in[4];
  const float* attn_b   = (const float*)d_in[5];
  const float* proj_w   = (const float*)d_in[6];
  const float* proj_b   = (const float*)d_in[7];
  const float* conv_w   = (const float*)d_in[8];
  const float* bn_g     = (const float*)d_in[9];
  const float* bn_b     = (const float*)d_in[10];
  const float* bn_m     = (const float*)d_in[11];
  const float* bn_v     = (const float*)d_in[12];
  const float* ln_m_g   = (const float*)d_in[13];
  const float* ln_m_b   = (const float*)d_in[14];
  const float* fc1_w    = (const float*)d_in[15];
  const float* fc1_b    = (const float*)d_in[16];
  const float* fc2_w    = (const float*)d_in[17];
  const float* fc2_b    = (const float*)d_in[18];
  const int*   bias_idx = (const int*)d_in[19];

  char* p = (char*)d_ws;
  auto carve = [&](size_t bytes) -> char* {
    char* r = p;
    p += (bytes + 255) & ~(size_t)255;
    return r;
  };
  _Float16* wq     = (_Float16*)carve((size_t)QKVO * C_ * 2);
  _Float16* wp     = (_Float16*)carve((size_t)C_ * C_ * 2);
  _Float16* w1     = (_Float16*)carve((size_t)HID_ * C_ * 2);
  _Float16* w2     = (_Float16*)carve((size_t)C_ * HID_ * 2);
  float*    bias12 = (float*)carve((size_t)NH_ * NTOK * NTOK * 4);
  _Float16* xn     = (_Float16*)carve((size_t)MROWS * C_ * 2);
  _Float16* qkvh   = (_Float16*)carve((size_t)MROWS * QKVO * 2);
  _Float16* Oh     = (_Float16*)carve((size_t)MROWS * C_ * 2);
  float*    x1     = (float*)carve((size_t)MROWS * C_ * 4);
  float*    x2     = (float*)carve((size_t)MROWS * C_ * 4);
  _Float16* xm     = (_Float16*)carve((size_t)MROWS * C_ * 2);
  _Float16* h1     = (_Float16*)carve((size_t)MROWS * HID_ * 2);

  // weight conversion + bias gather
  f32_to_f16_kernel<<<(QKVO * C_ + 255) / 256, 256, 0, stream>>>(qkv_w, wq, QKVO * C_);
  f32_to_f16_kernel<<<(C_ * C_ + 255) / 256, 256, 0, stream>>>(proj_w, wp, C_ * C_);
  f32_to_f16_kernel<<<(HID_ * C_ + 255) / 256, 256, 0, stream>>>(fc1_w, w1, HID_ * C_);
  f32_to_f16_kernel<<<(C_ * HID_ + 255) / 256, 256, 0, stream>>>(fc2_w, w2, C_ * HID_);
  bias_gather_kernel<<<dim3(NH_, (NTOK * NTOK + 255) / 256), 256, 0, stream>>>(
      attn_b, bias_idx, bias12);

  // window partition + LN -> f16
  ln_kernel<true><<<MROWS / 8, 256, 0, stream>>>(x, ln_a_g, ln_a_b, xn);

  // QKV GEMM: [50176,384] x [384,1152]
  gemm_wmma_kernel<0><<<dim3(QKVO / 128, MROWS / 128), 256, 0, stream>>>(
      xn, wq, qkv_b, nullptr, nullptr, qkvh, MROWS, QKVO, C_);

  // windowed attention
  attn_kernel<<<dim3(NH_, NWIN), 128, 0, stream>>>(qkvh, bias12, Oh);

  // proj GEMM + un-window + residual -> x1 (f32)
  gemm_wmma_kernel<2><<<dim3(C_ / 128, MROWS / 128), 256, 0, stream>>>(
      Oh, wp, proj_b, x, x1, nullptr, MROWS, C_, C_);

  // depthwise conv + BN -> x2
  dwconv_bn_kernel<<<MROWS, 384, 0, stream>>>(x1, conv_w, bn_g, bn_b, bn_m, bn_v, x2);

  // mid LN -> f16
  ln_kernel<false><<<MROWS / 8, 256, 0, stream>>>(x2, ln_m_g, ln_m_b, xm);

  // fc1 GEMM + GELU
  gemm_wmma_kernel<1><<<dim3(HID_ / 128, MROWS / 128), 256, 0, stream>>>(
      xm, w1, fc1_b, nullptr, nullptr, h1, MROWS, HID_, C_);

  // fc2 GEMM + residual -> d_out
  gemm_wmma_kernel<3><<<dim3(C_ / 128, MROWS / 128), 256, 0, stream>>>(
      h1, w2, fc2_b, x2, (float*)d_out, nullptr, MROWS, C_, HID_);
}
